// gcn_14259291422897
// MI455X (gfx1250) — matrix operand order repaired and checked by reading, not compiled, not run on hardware
//
#include <hip/hip_runtime.h>
#include <hip/hip_bf16.h>
#include <stdint.h>

typedef __bf16 bf16;
typedef bf16  v8bf  __attribute__((ext_vector_type(8)));
typedef bf16  v16bf __attribute__((ext_vector_type(16)));
typedef float v8f   __attribute__((ext_vector_type(8)));
typedef unsigned int u32x4 __attribute__((ext_vector_type(4)));
typedef int  i32x4 __attribute__((ext_vector_type(4)));
typedef int  i32x8 __attribute__((ext_vector_type(8)));

#define BB 64
#define CC 32
#define NN 500
#define LL 12
#define NP 512           // padded N (v and w padded with zeros)
#define CO 32
#define NG 9

#if defined(__gfx1250__) && __has_builtin(__builtin_amdgcn_tensor_load_to_lds)
#define HAVE_TDM 1
#else
#define HAVE_TDM 0
#endif

__device__ inline v16bf cat16(v8bf lo, v8bf hi) {
  return __builtin_shufflevector(lo, hi, 0,1,2,3,4,5,6,7,8,9,10,11,12,13,14,15);
}

__device__ inline v8f wmma_bf16(v16bf a, v16bf b, v8f c) {
  return __builtin_amdgcn_wmma_f32_16x16x32_bf16(false, a, false, b, (short)0, c, false, false);
}

#if HAVE_TDM
// 2-D TDM: load 32 rows x 512 bf16 (32 KB) from global (row stride 512 elems) to LDS.
__device__ inline void tdm_issue(const bf16* gsrc, unsigned lds_off) {
  unsigned long long ga = (unsigned long long)(uintptr_t)gsrc;
  u32x4 g0;
  g0[0] = 1u;                                  // count=1, user descriptor
  g0[1] = lds_off;                             // LDS byte address
  g0[2] = (unsigned)ga;                        // global_addr[31:0]
  g0[3] = (unsigned)(ga >> 32) | (2u << 30);   // global_addr[56:32] | type=2
  i32x8 g1;
  g1[0] = (int)(1u << 16);                     // data_size = 1 (2 bytes)
  g1[1] = (int)(512u << 16);                   // tensor_dim0 = 512 (lo16 at [31:16])
  g1[2] = (int)(32u << 16);                    // tensor_dim0 hi16=0 | tensor_dim1 = 32
  g1[3] = (int)(512u << 16);                   // tensor_dim1 hi16=0 | tile_dim0 = 512
  g1[4] = 32;                                  // tile_dim1 = 32, tile_dim2 = 0
  g1[5] = 512;                                 // tensor_dim0_stride = 512
  g1[6] = 0;
  g1[7] = 0;
  i32x4 z4 = {0, 0, 0, 0};
  i32x8 z8 = {0, 0, 0, 0, 0, 0, 0, 0};
  __builtin_amdgcn_tensor_load_to_lds(g0, g1, z4, z4, z8, 0);
}
#endif

// ---------------- prep: xT[b][c][l][v] = bf16(x[b][c][v][l]), v zero-padded to 512
__global__ void k_xT(const float* __restrict__ x, bf16* __restrict__ xT) {
  long tid = (long)blockIdx.x * 256 + threadIdx.x;   // BB*CC*LL*(NP/8) threads
  int v8 = (int)(tid % (NP / 8));
  long rest = tid / (NP / 8);
  int l = (int)(rest % LL); rest /= LL;
  int c = (int)(rest % CC);
  int b = (int)(rest / CC);
  if (b >= BB) return;
  int vb = v8 * 8;
  v8bf o;
#pragma unroll
  for (int j = 0; j < 8; ++j) {
    int v = vb + j;
    float f = (v < NN) ? x[(((long)(b * CC + c)) * NN + v) * LL + l] : 0.f;
    o[j] = (bf16)f;
  }
  *(v8bf*)&xT[(((long)(b * CC + c)) * LL + l) * NP + vb] = o;
}

// ---------------- prep: AT[m][w][v] = bf16(A[m][v][w]) padded to 512x512 (transposed)
__global__ void k_at(const float* __restrict__ A, bf16* __restrict__ AT, int batch) {
  long tid = (long)blockIdx.x * 256 + threadIdx.x;   // batch*NP*NP/8 threads
  int v8 = (int)(tid % (NP / 8));
  long rest = tid / (NP / 8);
  int w = (int)(rest % NP);
  int m = (int)(rest / NP);
  if (m >= batch) return;
  int vb = v8 * 8;
  v8bf o;
#pragma unroll
  for (int j = 0; j < 8; ++j) {
    int v = vb + j;
    float f = (v < NN && w < NN) ? A[((long)m * NN + v) * NN + w] : 0.f;
    o[j] = (bf16)f;
  }
  *(v8bf*)&AT[((long)m * NP + w) * NP + vb] = o;
}

// ---------------- prep: A16[m][v][u] = bf16(A[m][v][u]) padded to 512x512 (straight)
// coalesced reads (u fastest); zero padding removes all guards from k_sq's hot loop.
__global__ void k_a16(const float* __restrict__ A, bf16* __restrict__ A16, int batch) {
  long tid = (long)blockIdx.x * 256 + threadIdx.x;   // batch*NP*NP/8 threads
  int u8 = (int)(tid % (NP / 8));
  long rest = tid / (NP / 8);
  int v = (int)(rest % NP);
  int m = (int)(rest / NP);
  if (m >= batch) return;
  int ub = u8 * 8;
  v8bf o;
#pragma unroll
  for (int j = 0; j < 8; ++j) {
    int u = ub + j;
    float f = (u < NN && v < NN) ? A[((long)m * NN + v) * NN + u] : 0.f;
    o[j] = (bf16)f;
  }
  *(v8bf*)&A16[((long)m * NP + v) * NP + ub] = o;
}

// ---------------- prep: ATsq = AT x AT  (== A^2 stored transposed), WMMA bf16
// A-operand rows from AT (contiguous); B-operand B[k=u, n=v] = A[v][u] from A16 (contiguous).
__global__ __launch_bounds__(256) void k_sq(const bf16* __restrict__ AT,
                                            const bf16* __restrict__ A16,
                                            bf16* __restrict__ ATsq, int batch) {
  int lane = threadIdx.x & 31, wv = threadIdx.x >> 5;
  long flat = (long)blockIdx.x * 8 + wv;   // one 16x16 tile per wave
  int m = (int)(flat >> 10);
  if (m >= batch) return;
  int t = (int)(flat & 1023);
  int mt = t >> 5, nt = t & 31;
  int lm = lane & 15, hi = lane >> 4;
  int row = mt * 16 + lm;                  // w row
  int ncol = nt * 16 + lm;                 // v col
  const bf16* arow = AT  + ((long)m * NP + row)  * NP;
  const bf16* brow = A16 + ((long)m * NP + ncol) * NP + hi * 16;
  v8f acc = {};
  for (int kk = 0; kk < 16; ++kk) {
    int kb = kk * 32;
    v8bf a0 = *(const v8bf*)&arow[kb + hi * 16];
    v8bf a1 = *(const v8bf*)&arow[kb + hi * 16 + 8];
    v8bf b0 = *(const v8bf*)&brow[kb];
    v8bf b1 = *(const v8bf*)&brow[kb + 8];
    acc = wmma_bf16(cat16(a0, a1), cat16(b0, b1), acc);
  }
#pragma unroll
  for (int r = 0; r < 8; ++r) {
    int w = mt * 16 + r + hi * 8;
    ATsq[((long)m * NP + w) * NP + ncol] = (bf16)acc[r];
  }
}

// ---------------- prep: W pre-swizzled into WMMA B-operand lane order
// WB[g][ot][lane][k] = bf16( W[ot*16 + (lane&15)][g*32 + (lane>=16?16:0) + k] )
__global__ void k_wb(const float* __restrict__ W, bf16* __restrict__ WB) {
  int tid = blockIdx.x * 256 + threadIdx.x;
  if (tid >= NG * 2 * 32 * 16) return;
  int k = tid & 15;
  int p = (tid >> 4) & 31;
  int ot = (tid >> 9) & 1;
  int g = tid >> 10;
  int o = ot * 16 + (p & 15);
  int cc = g * 32 + ((p >> 4) << 4) + k;
  WB[tid] = (bf16)W[o * (NG * CC) + cc];
}

// ---------------- fused hop + channel-mix kernel
// grid: b(64) x wtile(16, 32 w each). 8 waves/block.
// Per group g: hop[w,(c,l)] = sum_v x[b,c,v,l]*Ag[v,w] via WMMA (K=v),
// round-trip through LDS as bf16, then out[w,(o,l)] += hop * W_g via WMMA (K=c).
__global__ __launch_bounds__(256, 1) void k_fused(
    const bf16* __restrict__ xT,
    const bf16* __restrict__ AT_s, const bf16* __restrict__ ATsq_s,
    const bf16* __restrict__ A3T,  const bf16* __restrict__ A3Tsq,
    const bf16* __restrict__ WB,   const float* __restrict__ bias,
    float* __restrict__ out) {
  extern __shared__ char smem[];
  bf16* aBuf   = (bf16*)smem;                       // [2][32][512] = 64 KB
  bf16* hopLDS = (bf16*)(smem + 2 * 32 * NP * 2);   // [12][32][32] = 24 KB
  const unsigned ldsBase = (unsigned)(uintptr_t)smem;
  (void)ldsBase;

  int wt = blockIdx.x & 15;
  int b  = blockIdx.x >> 4;
  int w0 = wt * 32;
  int lane = threadIdx.x & 31, wv = threadIdx.x >> 5;
  int mt = wv & 1;         // which 16-row M tile of the 32-wide w block
  int ns = wv >> 1;        // which 6 N tiles (of 24) this wave owns
  int lm = lane & 15, hi = lane >> 4;

  // adjacency slab base for group g (1..8), already transposed+padded
  auto slab = [&](int g) -> const bf16* {
    int q = (g - 1) >> 1;          // 0..3
    int sq = (g - 1) & 1;          // 0 = A, 1 = A^2
    if (q < 3) return (sq ? ATsq_s : AT_s) + (long)q * NP * NP;
    return (sq ? A3Tsq : A3T) + (long)b * NP * NP;
  };

  // B-operand row pointers (same for every group): col=(c,l), per-lane 16 consecutive v
  const bf16* xrow[6];
  int colc[6], coll[6];
#pragma unroll
  for (int i = 0; i < 6; ++i) {
    int colg = 16 * (ns * 6 + i) + lm;     // 0..383
    int c = colg / 12, l = colg - c * 12;
    colc[i] = c; coll[i] = l;
    xrow[i] = xT + (((long)(b * CC + c)) * LL + l) * NP + hi * 16;
  }

  v8f outAcc[6];
#pragma unroll
  for (int i = 0; i < 6; ++i) outAcc[i] = (v8f){};

#if HAVE_TDM
  if (wv == 0) tdm_issue(slab(1) + (long)w0 * NP, ldsBase + 32768u);
#endif

  for (int g = 0; g < 9; ++g) {
    if (g == 0) {
      // identity group: stage x tile straight into hopLDS[l][w][c]
#pragma unroll 4
      for (int i = 0; i < 48; ++i) {
        int flat = threadIdx.x + 256 * i;
        int c = flat & 31, w = (flat >> 5) & 31, l = flat >> 10;
        hopLDS[(l * 32 + w) * 32 + c] =
            xT[(((long)(b * CC + c)) * LL + l) * NP + w0 + w];
      }
    } else {
#if HAVE_TDM
      if (wv == 0) __builtin_amdgcn_s_wait_tensorcnt(0);
      __syncthreads();                                  // slab(g) visible to all
      if (g < 8 && wv == 0)
        tdm_issue(slab(g + 1) + (long)w0 * NP, ldsBase + ((unsigned)((g + 1) & 1)) * 32768u);
#else
      __syncthreads();
      {
        const bf16* src = slab(g) + (long)w0 * NP;
        v8bf* dst = (v8bf*)(aBuf + (g & 1) * 32 * NP);
#pragma unroll
        for (int i = 0; i < 8; ++i)
          dst[threadIdx.x + 256 * i] = *(const v8bf*)&src[(threadIdx.x + 256 * i) * 8];
      }
      __syncthreads();
#endif
      const bf16* abase = aBuf + ((long)((g & 1) * 32 + mt * 16 + lm)) * NP;
      v8f hop[6];
#pragma unroll
      for (int i = 0; i < 6; ++i) hop[i] = (v8f){};
      for (int kk = 0; kk < 16; ++kk) {
        int kb = kk * 32;
        v8bf a0 = *(const v8bf*)&abase[kb + hi * 16];
        v8bf a1 = *(const v8bf*)&abase[kb + hi * 16 + 8];
        v16bf av = cat16(a0, a1);
#pragma unroll
        for (int i = 0; i < 6; ++i) {
          v8bf b0 = *(const v8bf*)&xrow[i][kb];
          v8bf b1 = *(const v8bf*)&xrow[i][kb + 8];
          hop[i] = wmma_bf16(av, cat16(b0, b1), hop[i]);
        }
      }
      // spill hop tiles to hopLDS[l][w][c] as bf16
#pragma unroll
      for (int i = 0; i < 6; ++i) {
        int c = colc[i], l = coll[i];
#pragma unroll
        for (int r = 0; r < 8; ++r) {
          int wl = mt * 16 + r + hi * 8;
          hopLDS[(l * 32 + wl) * 32 + c] = (bf16)hop[i][r];
        }
      }
    }
    __syncthreads();   // hopLDS ready
    // mix: out[w,(o,l)] += hop[w,c,l] * W_g[o,c]   (K = c = 32, one WMMA per tile)
#pragma unroll
    for (int i = 0; i < 6; ++i) {
      int j = ns * 6 + i;
      int ot = j / 12, l = j - ot * 12;
      const bf16* hrow = &hopLDS[(l * 32 + mt * 16 + lm) * 32];
      v8bf a0 = *(const v8bf*)&hrow[hi * 16];
      v8bf a1 = *(const v8bf*)&hrow[hi * 16 + 8];
      const bf16* wrow = WB + ((long)(g * 2 + ot) * 32 + lane) * 16;
      v8bf b0 = *(const v8bf*)&wrow[0];
      v8bf b1 = *(const v8bf*)&wrow[8];
      outAcc[i] = wmma_bf16(cat16(a0, a1), cat16(b0, b1), outAcc[i]);
    }
    __syncthreads();   // done reading hopLDS before next group overwrites it
  }

  // epilogue: bias + store fp32 out[b][o][w][l]
#pragma unroll
  for (int i = 0; i < 6; ++i) {
    int j = ns * 6 + i;
    int ot = j / 12, l = j - ot * 12;
    int o = ot * 16 + lm;
    float bv = bias[o];
#pragma unroll
    for (int r = 0; r < 8; ++r) {
      int w = w0 + mt * 16 + r + hi * 8;
      if (w < NN)
        out[(((long)(b * CO + o)) * NN + w) * LL + l] = outAcc[i][r] + bv;
    }
  }
}

extern "C" void kernel_launch(void* const* d_in, const int* in_sizes, int n_in,
                              void* d_out, int out_size, void* d_ws, size_t ws_size,
                              hipStream_t stream) {
  const float* x  = (const float*)d_in[0];
  const float* A0 = (const float*)d_in[1];
  const float* A1 = (const float*)d_in[2];
  const float* A2 = (const float*)d_in[3];
  const float* A3 = (const float*)d_in[4];
  const float* W  = (const float*)d_in[5];
  const float* bb = (const float*)d_in[6];
  float* out = (float*)d_out;

  char* ws = (char*)d_ws;
  size_t off = 0;
  auto alloc = [&](size_t bytes) -> void* {
    void* p = ws + off;
    off = (off + bytes + 255) & ~(size_t)255;
    return p;
  };
  bf16* xT     = (bf16*)alloc((size_t)BB * CC * LL * NP * 2);   // 25.2 MB
  bf16* AT_s   = (bf16*)alloc((size_t)3 * NP * NP * 2);
  bf16* ATsq_s = (bf16*)alloc((size_t)3 * NP * NP * 2);
  bf16* A16_s  = (bf16*)alloc((size_t)3 * NP * NP * 2);
  bf16* A3T    = (bf16*)alloc((size_t)BB * NP * NP * 2);        // 33.6 MB
  bf16* A3Tsq  = (bf16*)alloc((size_t)BB * NP * NP * 2);        // 33.6 MB
  bf16* A316   = (bf16*)alloc((size_t)BB * NP * NP * 2);        // 33.6 MB
  bf16* WB     = (bf16*)alloc((size_t)NG * 2 * 32 * 16 * 2);

  // prep: bf16 conversions (transposed + straight, both zero-padded to 512x512)
  k_xT<<<dim3((BB * CC * LL * (NP / 8)) / 256), dim3(256), 0, stream>>>(x, xT);
  k_at<<<dim3(128), dim3(256), 0, stream>>>(A0, AT_s + 0 * NP * NP, 1);
  k_at<<<dim3(128), dim3(256), 0, stream>>>(A1, AT_s + 1 * NP * NP, 1);
  k_at<<<dim3(128), dim3(256), 0, stream>>>(A2, AT_s + 2 * NP * NP, 1);
  k_at<<<dim3(64 * 128), dim3(256), 0, stream>>>(A3, A3T, 64);
  k_a16<<<dim3(128), dim3(256), 0, stream>>>(A0, A16_s + 0 * NP * NP, 1);
  k_a16<<<dim3(128), dim3(256), 0, stream>>>(A1, A16_s + 1 * NP * NP, 1);
  k_a16<<<dim3(128), dim3(256), 0, stream>>>(A2, A16_s + 2 * NP * NP, 1);
  k_a16<<<dim3(64 * 128), dim3(256), 0, stream>>>(A3, A316, 64);
  // squares (A^2, stored transposed) via WMMA
  k_sq<<<dim3(128), dim3(256), 0, stream>>>(AT_s + 0 * NP * NP, A16_s + 0 * NP * NP,
                                            ATsq_s + 0 * NP * NP, 1);
  k_sq<<<dim3(128), dim3(256), 0, stream>>>(AT_s + 1 * NP * NP, A16_s + 1 * NP * NP,
                                            ATsq_s + 1 * NP * NP, 1);
  k_sq<<<dim3(128), dim3(256), 0, stream>>>(AT_s + 2 * NP * NP, A16_s + 2 * NP * NP,
                                            ATsq_s + 2 * NP * NP, 1);
  k_sq<<<dim3(64 * 128), dim3(256), 0, stream>>>(A3T, A316, A3Tsq, 64);
  k_wb<<<dim3(36), dim3(256), 0, stream>>>(W, WB);

  // fused hop + mix (88 KB dynamic LDS: 2x32KB A-slab double buffer + 24KB hop tile)
  k_fused<<<dim3(BB * 16), dim3(256), 90112, stream>>>(
      xT, AT_s, ATsq_s, A3T, A3Tsq, WB, bb, out);
}
